// TransformerBlock_49374944034842
// MI455X (gfx1250) — matrix operand order repaired and checked by reading, not compiled, not run on hardware
//
#include <hip/hip_runtime.h>

// ---------------------------------------------------------------------------
// Transformer block for MI455X (gfx1250, wave32, WMMA).
// GEMMs + attention run in BF16 with FP32 accumulation via
// v_wmma_f32_16x16x32_bf16; LayerNorm / softmax / residuals stay FP32.
// GEMM uses double-buffered LDS with GLOBAL_LOAD_ASYNC_TO_LDS (ASYNCcnt):
// tile t+1 DMAs while tile t is in the WMMA pipe; one barrier per K-step.
// ---------------------------------------------------------------------------

typedef __attribute__((ext_vector_type(16))) __bf16 v16bf;
typedef __attribute__((ext_vector_type(8)))  float  v8f;
typedef __attribute__((__vector_size__(4 * sizeof(int)))) int v4i;
typedef unsigned short u16;

union Frag16 { uint4 u[2]; v16bf v; u16 h[16]; };

#if defined(__AMDGCN__) && __has_builtin(__builtin_amdgcn_global_load_async_to_lds_b128)
#define HAS_ASYNC 1
typedef __attribute__((address_space(1))) v4i* gv4i_p;
typedef __attribute__((address_space(3))) v4i* lv4i_p;
__device__ inline void async_cp16(const void* g, void* l) {
    __builtin_amdgcn_global_load_async_to_lds_b128(
        (gv4i_p)(v4i*)(void*)g,
        (lv4i_p)(v4i*)(void*)l, 0, 0);
}
__device__ inline void wait_async() {
#if __has_builtin(__builtin_amdgcn_s_wait_asynccnt)
    __builtin_amdgcn_s_wait_asynccnt(0);
#else
    asm volatile("s_wait_asynccnt 0x0" ::: "memory");
#endif
}
#else
#define HAS_ASYNC 0
#endif

__device__ inline u16 f2bf(float f) {
    unsigned int u = __float_as_uint(f);
    unsigned int r = (u + 0x7FFFu + ((u >> 16) & 1u)) >> 16;
    return (u16)r;
}

__device__ inline float gelu_tanh(float x) {
    float x3 = x * x * x;
    return 0.5f * x * (1.f + tanhf(0.79788456080286535588f * (x + 0.044715f * x3)));
}

__device__ inline float halfmax16(float v) {
    #pragma unroll
    for (int m = 1; m < 16; m <<= 1) v = fmaxf(v, __shfl_xor(v, m, 32));
    return v;
}
__device__ inline float halfsum16(float v) {
    #pragma unroll
    for (int m = 1; m < 16; m <<= 1) v += __shfl_xor(v, m, 32);
    return v;
}

// ---------------------------------------------------------------------------
// fp32 -> bf16 conversion (weights)
// ---------------------------------------------------------------------------
__global__ __launch_bounds__(256) void cvt_bf16_kernel(const float* __restrict__ in,
                                                       u16* __restrict__ out, int n) {
    int i = blockIdx.x * 256 + threadIdx.x;
    if (i < n) out[i] = f2bf(in[i]);
}

// ---------------------------------------------------------------------------
// LayerNorm (fp32 in) -> bf16 out.  One row (D=1024) per 256-thread block.
// ---------------------------------------------------------------------------
__global__ __launch_bounds__(256) void ln_bf16_kernel(const float* __restrict__ x,
                                                      const float* __restrict__ g,
                                                      const float* __restrict__ b,
                                                      u16* __restrict__ out, int D) {
    int row = blockIdx.x;
    const float* xr = x + (size_t)row * D;
    float s = 0.f, sq = 0.f;
    for (int i = threadIdx.x; i < D; i += 256) { float v = xr[i]; s += v; sq += v * v; }
    #pragma unroll
    for (int m = 1; m < 32; m <<= 1) { s += __shfl_xor(s, m, 32); sq += __shfl_xor(sq, m, 32); }
    __shared__ float ss[8], qq[8];
    int w = threadIdx.x >> 5, ln = threadIdx.x & 31;
    if (ln == 0) { ss[w] = s; qq[w] = sq; }
    __syncthreads();
    if (threadIdx.x == 0) {
        float ts = 0.f, tq = 0.f;
        for (int i = 0; i < 8; ++i) { ts += ss[i]; tq += qq[i]; }
        ss[0] = ts; qq[0] = tq;
    }
    __syncthreads();
    float mu   = ss[0] / (float)D;
    float var  = qq[0] / (float)D - mu * mu;
    float rstd = rsqrtf(var + 1e-6f);
    for (int i = threadIdx.x; i < D; i += 256)
        out[(size_t)row * D + i] = f2bf((xr[i] - mu) * rstd * g[i] + b[i]);
}

// ---------------------------------------------------------------------------
// BF16 GEMM: C[M,N] = epi(A[M,K] @ W[K,N]).  Block tile 128x64x32, 8 waves,
// each wave owns a 32x32 output tile (2x2 WMMA 16x16 frags).
// Double-buffered LDS:
//   sA[2]: A row-major [128][40] halfs (80B rows, 16B aligned)
//   sB[2]: W^T tile [64 n][40 k] halfs (B-fragment reads contiguous)
// Pipeline per K-step: issue async A(t+1) + global-load W(t+1) -> WMMA on
// tile t -> scatter W(t+1) -> s_wait_asynccnt -> barrier.
// ---------------------------------------------------------------------------
template <bool BIAS, bool GELU, bool RES, bool OUT_BF>
__global__ __launch_bounds__(256)
void gemm_bf16_kernel(const u16* __restrict__ A, const u16* __restrict__ W,
                      const float* __restrict__ bias, const float* __restrict__ res,
                      void* __restrict__ out, int M, int N, int K) {
    __shared__ __align__(16) u16 sA[2][128 * 40];
    __shared__ __align__(16) u16 sB[2][64 * 40];

    const int tid    = threadIdx.x;
    const int wave   = tid >> 5;
    const int lane   = tid & 31;
    const int laneLo = lane & 15;
    const int laneHi = lane >> 4;
    const int wm     = wave >> 1;          // 0..3 -> 32-row strip
    const int wn     = wave & 1;           // 0..1 -> 32-col strip
    const int m0     = blockIdx.y * 128;
    const int n0     = blockIdx.x * 64;
    const int krow   = tid >> 3, bseg = tid & 7;   // 32 k-rows x 8 segs

    v8f acc[2][2];
    #pragma unroll
    for (int i = 0; i < 2; ++i)
        #pragma unroll
        for (int j = 0; j < 2; ++j)
            #pragma unroll
            for (int r = 0; r < 8; ++r) acc[i][j][r] = 0.f;

    // ---- prologue: stage tile 0 into buffer 0 ----
    {
#if HAS_ASYNC
        #pragma unroll
        for (int i = 0; i < 2; ++i) {
            int c = tid + i * 256;                 // 512 chunks, 128 rows x 4
            int r = c >> 2, sgi = c & 3;
            async_cp16(A + (size_t)(m0 + r) * K + sgi * 8, &sA[0][r * 40 + sgi * 8]);
        }
#else
        #pragma unroll
        for (int i = 0; i < 2; ++i) {
            int c = tid + i * 256;
            int r = c >> 2, sgi = c & 3;
            *(uint4*)&sA[0][r * 40 + sgi * 8] =
                *(const uint4*)(A + (size_t)(m0 + r) * K + sgi * 8);
        }
#endif
        uint4 b0 = *(const uint4*)(W + (size_t)krow * N + n0 + bseg * 8);
        union { uint4 u; u16 h[8]; } cv; cv.u = b0;
        #pragma unroll
        for (int jj = 0; jj < 8; ++jj)
            sB[0][(bseg * 8 + jj) * 40 + krow] = cv.h[jj];
#if HAS_ASYNC
        wait_async();
#endif
        __syncthreads();
    }

    const int nT = K >> 5;
    for (int t = 0; t < nT; ++t) {
        const int cur = t & 1, nxt = cur ^ 1;
        const bool haveNext = (t + 1) < nT;
        const int ktn = (t + 1) << 5;

        // ---- kick off staging of tile t+1 (overlaps WMMA below) ----
        uint4 bReg;
#if !HAS_ASYNC
        uint4 aReg[2];
#endif
        if (haveNext) {
#if HAS_ASYNC
            #pragma unroll
            for (int i = 0; i < 2; ++i) {
                int c = tid + i * 256;
                int r = c >> 2, sgi = c & 3;
                async_cp16(A + (size_t)(m0 + r) * K + ktn + sgi * 8,
                           &sA[nxt][r * 40 + sgi * 8]);
            }
#else
            #pragma unroll
            for (int i = 0; i < 2; ++i) {
                int c = tid + i * 256;
                int r = c >> 2, sgi = c & 3;
                aReg[i] = *(const uint4*)(A + (size_t)(m0 + r) * K + ktn + sgi * 8);
            }
#endif
            bReg = *(const uint4*)(W + (size_t)(ktn + krow) * N + n0 + bseg * 8);
            if (t + 2 < nT) {                      // warm GL2 two K-steps ahead
                __builtin_prefetch(W + (size_t)(ktn + 32 + krow) * N + n0 + bseg * 8, 0, 1);
                __builtin_prefetch(A + (size_t)(m0 + (tid >> 1)) * K + ktn + 32 + (tid & 1) * 16, 0, 1);
            }
        }

        // ---- compute tile t (fragments per documented CDNA5 16-bit layouts) ----
        Frag16 aF[2], bF[2];
        #pragma unroll
        for (int s2 = 0; s2 < 2; ++s2) {
            int m = wm * 32 + s2 * 16 + laneLo;
            aF[s2].u[0] = *(const uint4*)&sA[cur][m * 40 + laneHi * 8];
            aF[s2].u[1] = *(const uint4*)&sA[cur][m * 40 + laneHi * 8 + 16];
            int n = wn * 32 + s2 * 16 + laneLo;
            bF[s2].u[0] = *(const uint4*)&sB[cur][n * 40 + laneHi * 8];
            bF[s2].u[1] = *(const uint4*)&sB[cur][n * 40 + laneHi * 8 + 16];
        }
        #pragma unroll
        for (int sm = 0; sm < 2; ++sm)
            #pragma unroll
            for (int sn = 0; sn < 2; ++sn)
                acc[sm][sn] = __builtin_amdgcn_wmma_f32_16x16x32_bf16(
                    false, aF[sm].v, false, bF[sn].v, (short)0, acc[sm][sn], false, false);

        // ---- finish staging tile t+1, sync once ----
        if (haveNext) {
            union { uint4 u; u16 h[8]; } cv; cv.u = bReg;
            #pragma unroll
            for (int jj = 0; jj < 8; ++jj)
                sB[nxt][(bseg * 8 + jj) * 40 + krow] = cv.h[jj];
#if !HAS_ASYNC
            #pragma unroll
            for (int i = 0; i < 2; ++i) {
                int c = tid + i * 256;
                int r = c >> 2, sgi = c & 3;
                *(uint4*)&sA[nxt][r * 40 + sgi * 8] = aReg[i];
            }
#else
            wait_async();
#endif
            __syncthreads();
        }
    }

    // epilogue: C layout -> lane holds col n0+..+laneLo, rows r (+8 if laneHi)
    #pragma unroll
    for (int sm = 0; sm < 2; ++sm)
        #pragma unroll
        for (int sn = 0; sn < 2; ++sn) {
            int col = n0 + wn * 32 + sn * 16 + laneLo;
            float bv = 0.f;
            if constexpr (BIAS) bv = bias[col];
            #pragma unroll
            for (int r = 0; r < 8; ++r) {
                int row = m0 + wm * 32 + sm * 16 + laneHi * 8 + r;
                float v = acc[sm][sn][r] + bv;
                if constexpr (GELU) v = gelu_tanh(v);
                if constexpr (RES)  v += res[(size_t)row * N + col];
                if constexpr (OUT_BF) ((u16*)out)[(size_t)row * N + col] = f2bf(v);
                else                  ((float*)out)[(size_t)row * N + col] = v;
            }
        }
}

// ---------------------------------------------------------------------------
// Flash-style causal attention.  H=16 heads, HS=64.  qkv is bf16 [B,N,3D].
// Block = 128 query rows of one (b,h); 8 waves, each owns 16 rows.
// Per 64-key tile: S = Q K^T (8 WMMA), online softmax, P.V (8 WMMA).
// Q/K tiles: async HBM->LDS.  V tile: transposed via reg+ds scatter.
// ---------------------------------------------------------------------------
__global__ __launch_bounds__(256)
void attn_kernel(const u16* __restrict__ qkv, u16* __restrict__ outh) {
    const int N = 2048, TD = 3072, Dm = 1024;
    const int qt = blockIdx.x, h = blockIdx.y, b = blockIdx.z;
    const int tid = threadIdx.x, wave = tid >> 5, lane = tid & 31;
    const int laneLo = lane & 15, laneHi = lane >> 4;
    const int qBase = qt * 128;

    const u16* qkvB = qkv + (size_t)b * N * TD;
    const u16* Qp = qkvB + h * 64;
    const u16* Kp = qkvB + Dm + h * 64;
    const u16* Vp = qkvB + 2 * Dm + h * 64;

    __shared__ __align__(16) u16 sQ[128 * 72];
    __shared__ __align__(16) u16 sK[64 * 72];      // [key][hs]  == B^T for QK^T
    __shared__ __align__(16) u16 sVt[64 * 72];     // [hs][key]  == B^T for P.V
    __shared__ __align__(16) u16 sP[8][16 * 72];   // per-wave P tile

    // load Q tile (128x64 halfs, 16B chunks, coalesced)
#if HAS_ASYNC
    #pragma unroll
    for (int i = 0; i < 4; ++i) {
        int c = tid + i * 256;                     // 1024 chunks
        int r = c >> 3, sg = c & 7;
        async_cp16(Qp + (size_t)(qBase + r) * TD + sg * 8, &sQ[r * 72 + sg * 8]);
    }
    wait_async();
#else
    #pragma unroll
    for (int i = 0; i < 4; ++i) {
        int c = tid + i * 256;
        int r = c >> 3, sg = c & 7;
        uint4 v = *(const uint4*)(Qp + (size_t)(qBase + r) * TD + sg * 8);
        *(uint4*)&sQ[r * 72 + sg * 8] = v;
    }
#endif
    __syncthreads();

    Frag16 aQ[2];
    {
        int m = wave * 16 + laneLo;
        #pragma unroll
        for (int kk = 0; kk < 2; ++kk) {
            aQ[kk].u[0] = *(const uint4*)&sQ[m * 72 + kk * 32 + laneHi * 8];
            aQ[kk].u[1] = *(const uint4*)&sQ[m * 72 + kk * 32 + laneHi * 8 + 16];
        }
    }

    float mrun[8], lrun[8];
    v8f oacc[4];
    #pragma unroll
    for (int r = 0; r < 8; ++r) { mrun[r] = -3.0e38f; lrun[r] = 0.f; }
    #pragma unroll
    for (int t = 0; t < 4; ++t)
        #pragma unroll
        for (int r = 0; r < 8; ++r) oacc[t][r] = 0.f;

    const int jEnd = (qBase + 128) >> 6;
    for (int j = 0; j < jEnd; ++j) {
        // stage K (async, direct) and V (reg + transpose scatter) tiles
        uint4 vReg[2];
#if !HAS_ASYNC
        uint4 kReg[2];
#endif
        int rw[2], sg2[2];
        #pragma unroll
        for (int i = 0; i < 2; ++i) {
            int c = tid + i * 256;                 // 512 chunks
            rw[i] = c >> 3; sg2[i] = c & 7;
#if !HAS_ASYNC
            kReg[i] = *(const uint4*)(Kp + (size_t)(j * 64 + rw[i]) * TD + sg2[i] * 8);
#endif
            vReg[i] = *(const uint4*)(Vp + (size_t)(j * 64 + rw[i]) * TD + sg2[i] * 8);
        }
        if (j + 1 < jEnd) {                        // warm GL2 one tile ahead
            __builtin_prefetch(Kp + (size_t)((j + 1) * 64 + rw[0]) * TD + sg2[0] * 8, 0, 1);
            __builtin_prefetch(Vp + (size_t)((j + 1) * 64 + rw[0]) * TD + sg2[0] * 8, 0, 1);
        }
        __syncthreads();
#if HAS_ASYNC
        #pragma unroll
        for (int i = 0; i < 2; ++i)
            async_cp16(Kp + (size_t)(j * 64 + rw[i]) * TD + sg2[i] * 8,
                       &sK[rw[i] * 72 + sg2[i] * 8]);
#else
        #pragma unroll
        for (int i = 0; i < 2; ++i)
            *(uint4*)&sK[rw[i] * 72 + sg2[i] * 8] = kReg[i];
#endif
        #pragma unroll
        for (int i = 0; i < 2; ++i) {
            union { uint4 u; u16 hh[8]; } cv; cv.u = vReg[i];
            #pragma unroll
            for (int jj = 0; jj < 8; ++jj)
                sVt[(sg2[i] * 8 + jj) * 72 + rw[i]] = cv.hh[jj];
        }
#if HAS_ASYNC
        wait_async();
#endif
        __syncthreads();

        // S = Q K^T  (16 q-rows x 64 keys per wave)
        v8f sacc[4];
        #pragma unroll
        for (int nt = 0; nt < 4; ++nt)
            #pragma unroll
            for (int r = 0; r < 8; ++r) sacc[nt][r] = 0.f;
        #pragma unroll
        for (int nt = 0; nt < 4; ++nt) {
            int n = nt * 16 + laneLo;
            #pragma unroll
            for (int kk = 0; kk < 2; ++kk) {
                Frag16 bK;
                bK.u[0] = *(const uint4*)&sK[n * 72 + kk * 32 + laneHi * 8];
                bK.u[1] = *(const uint4*)&sK[n * 72 + kk * 32 + laneHi * 8 + 16];
                sacc[nt] = __builtin_amdgcn_wmma_f32_16x16x32_bf16(
                    false, aQ[kk].v, false, bK.v, (short)0, sacc[nt], false, false);
            }
        }

        // scale + causal mask + online softmax (rows split across laneHi halves)
        const float scale = 0.125f;                // rsqrt(64)
        const bool diagTile = (j * 64 + 63) > (qBase + wave * 16);
        float p[4][8];
        #pragma unroll
        for (int r = 0; r < 8; ++r) {
            int q = qBase + wave * 16 + laneHi * 8 + r;
            float mx = -3.0e38f;
            #pragma unroll
            for (int nt = 0; nt < 4; ++nt) {
                float v = sacc[nt][r] * scale;
                int key = j * 64 + nt * 16 + laneLo;
                if (diagTile && key > q) v = -3.0e38f;
                p[nt][r] = v;
                mx = fmaxf(mx, v);
            }
            mx = halfmax16(mx);
            float mnew  = fmaxf(mrun[r], mx);
            float alpha = __expf(mrun[r] - mnew);
            float ls = 0.f;
            #pragma unroll
            for (int nt = 0; nt < 4; ++nt) {
                float e = __expf(p[nt][r] - mnew);
                p[nt][r] = e;
                ls += e;
            }
            ls = halfsum16(ls);
            mrun[r] = mnew;
            lrun[r] = lrun[r] * alpha + ls;
            #pragma unroll
            for (int nt = 0; nt < 4; ++nt) oacc[nt][r] *= alpha;
        }

        // C-layout P -> per-wave LDS (row-major) -> A-layout fragments
        #pragma unroll
        for (int nt = 0; nt < 4; ++nt)
            #pragma unroll
            for (int r = 0; r < 8; ++r)
                sP[wave][(laneHi * 8 + r) * 72 + nt * 16 + laneLo] = f2bf(p[nt][r]);

        Frag16 aP[2];
        #pragma unroll
        for (int kk = 0; kk < 2; ++kk) {
            aP[kk].u[0] = *(const uint4*)&sP[wave][laneLo * 72 + kk * 32 + laneHi * 8];
            aP[kk].u[1] = *(const uint4*)&sP[wave][laneLo * 72 + kk * 32 + laneHi * 8 + 16];
        }
        #pragma unroll
        for (int nt = 0; nt < 4; ++nt) {
            int n = nt * 16 + laneLo;
            #pragma unroll
            for (int kk = 0; kk < 2; ++kk) {
                Frag16 bV;
                bV.u[0] = *(const uint4*)&sVt[n * 72 + kk * 32 + laneHi * 8];
                bV.u[1] = *(const uint4*)&sVt[n * 72 + kk * 32 + laneHi * 8 + 16];
                oacc[nt] = __builtin_amdgcn_wmma_f32_16x16x32_bf16(
                    false, aP[kk].v, false, bV.v, (short)0, oacc[nt], false, false);
            }
        }
    }

    // normalize and write merged-head output (bf16)
    #pragma unroll
    for (int nt = 0; nt < 4; ++nt)
        #pragma unroll
        for (int r = 0; r < 8; ++r) {
            int q = qBase + wave * 16 + laneHi * 8 + r;
            int d = h * 64 + nt * 16 + laneLo;
            outh[((size_t)b * N + q) * Dm + d] = f2bf(oacc[nt][r] / lrun[r]);
        }
}

// ---------------------------------------------------------------------------
// Host-side orchestration
// ---------------------------------------------------------------------------
extern "C" void kernel_launch(void* const* d_in, const int* in_sizes, int n_in,
                              void* d_out, int out_size, void* d_ws, size_t ws_size,
                              hipStream_t stream) {
    const float* x    = (const float*)d_in[0];
    const float* g1   = (const float*)d_in[1];
    const float* bb1  = (const float*)d_in[2];
    const float* wqkv = (const float*)d_in[3];
    const float* wout = (const float*)d_in[4];
    const float* bout = (const float*)d_in[5];
    const float* g2   = (const float*)d_in[6];
    const float* bb2  = (const float*)d_in[7];
    const float* w1   = (const float*)d_in[8];
    const float* b1   = (const float*)d_in[9];
    const float* w2   = (const float*)d_in[10];
    const float* b2   = (const float*)d_in[11];
    (void)in_sizes; (void)n_in; (void)out_size; (void)ws_size;

    const int M = 4096;   // B*N tokens
    const int D = 1024;

    char* p = (char*)d_ws;
    auto take = [&](size_t bytes) { char* q = p; p += (bytes + 255) & ~(size_t)255; return q; };
    u16*   wqkv_bf = (u16*)take((size_t)D * 3 * D * 2);
    u16*   wout_bf = (u16*)take((size_t)D * D * 2);
    u16*   w1_bf   = (u16*)take((size_t)D * 2 * D * 2);
    u16*   w2_bf   = (u16*)take((size_t)2 * D * D * 2);
    u16*   xn_bf   = (u16*)take((size_t)M * D * 2);        // reused for ln1 & ln2
    u16*   qkv_bf  = (u16*)take((size_t)M * 3 * D * 2);
    u16*   ao_bf   = (u16*)take((size_t)M * D * 2);
    float* x1      = (float*)take((size_t)M * D * 4);
    u16*   h_bf    = (u16*)take((size_t)M * 2 * D * 2);

    // weights -> bf16
    cvt_bf16_kernel<<<(D * 3 * D + 255) / 256, 256, 0, stream>>>(wqkv, wqkv_bf, D * 3 * D);
    cvt_bf16_kernel<<<(D * D + 255) / 256, 256, 0, stream>>>(wout, wout_bf, D * D);
    cvt_bf16_kernel<<<(D * 2 * D + 255) / 256, 256, 0, stream>>>(w1, w1_bf, D * 2 * D);
    cvt_bf16_kernel<<<(2 * D * D + 255) / 256, 256, 0, stream>>>(w2, w2_bf, 2 * D * D);

    // LN1, QKV projection
    ln_bf16_kernel<<<M, 256, 0, stream>>>(x, g1, bb1, xn_bf, D);
    gemm_bf16_kernel<false, false, false, true>
        <<<dim3(3 * D / 64, M / 128), 256, 0, stream>>>(xn_bf, wqkv_bf, nullptr, nullptr,
                                                        qkv_bf, M, 3 * D, D);
    // attention
    attn_kernel<<<dim3(16, 16, 2), 256, 0, stream>>>(qkv_bf, ao_bf);

    // out projection + residual (fp32)
    gemm_bf16_kernel<true, false, true, false>
        <<<dim3(D / 64, M / 128), 256, 0, stream>>>(ao_bf, wout_bf, bout, x, x1, M, D, D);

    // LN2, MLP
    ln_bf16_kernel<<<M, 256, 0, stream>>>(x1, g2, bb2, xn_bf, D);
    gemm_bf16_kernel<true, true, false, true>
        <<<dim3(2 * D / 64, M / 128), 256, 0, stream>>>(xn_bf, w1_bf, b1, nullptr,
                                                        h_bf, M, 2 * D, D);
    gemm_bf16_kernel<true, false, true, false>
        <<<dim3(D / 64, M / 128), 256, 0, stream>>>(h_bf, w2_bf, b2, x1,
                                                    (float*)d_out, M, D, 2 * D);
}
